// GNNModel_83279415869582
// MI455X (gfx1250) — matrix operand order repaired and checked by reading, not compiled, not run on hardware
//
#include <hip/hip_runtime.h>

#define N_NODES  50000
#define N_EDGES  800000
#define N_GRAPHS 2048
#define HID      128
#define BN_EPS   1e-5f

typedef __attribute__((ext_vector_type(16))) __bf16 v16bf;
typedef __attribute__((ext_vector_type(8)))  float  v8f;

// exact RNE scalar conversion (cold paths: weight packing)
__device__ __forceinline__ unsigned short f2bf(float f) {
    unsigned int u = __float_as_uint(f);
    u += 0x7FFFu + ((u >> 16) & 1u);
    return (unsigned short)(u >> 16);
}

// hot-path packed f32x2 -> bf16x2:
//   1) native V_CVT_PK_BF16_F32 if the builtin exists (not in current toolchain)
//   2) else round-to-nearest (+0x8000) + single V_PERM_B32 to pack both high halves
__device__ __forceinline__ void pk2bf(unsigned short* dst, float f0, float f1) {
#if __has_builtin(__builtin_amdgcn_cvt_pk_bf16_f32)
    auto p = __builtin_amdgcn_cvt_pk_bf16_f32(f0, f1);
    __builtin_memcpy(dst, &p, 4);
#else
    unsigned int r0 = __float_as_uint(f0) + 0x8000u;
    unsigned int r1 = __float_as_uint(f1) + 0x8000u;
    // dst = {hi16(r1), hi16(r0)} : sel bytes {7,6,3,2} over {S0=r1, S1=r0}
    unsigned int p = __builtin_amdgcn_perm(r1, r0, 0x07060302u);
    __builtin_memcpy(dst, &p, 4);
#endif
}

union BFVec { v16bf v; unsigned short u[16]; uint4 q[2]; };

__device__ __forceinline__ void cvt4(unsigned short* dst, float4 v) {
    pk2bf(dst, v.x, v.y);
    pk2bf(dst + 2, v.z, v.w);
}

// ---------------------------------------------------------------------------
// Pack W [128, NCOL] (f32 row-major) into bf16 WMMA B-operand layout:
// Wp[((colTile*4 + ks)*32 + lane)*16 + j] = bf16(W[ks*32 + (j>>3)*16 + (lane>>4)*8 + (j&7)][colTile*16 + (lane&15)])
// -> in the GEMM each lane fetches its 16 bf16 B values as two b128 loads.
// ---------------------------------------------------------------------------
template<int NCOL>
__global__ void pack_w_bf16(const float* __restrict__ W, unsigned short* __restrict__ Wp) {
    int idx = blockIdx.x * blockDim.x + threadIdx.x;
    if (idx >= NCOL * 128) return;
    int j       = idx & 15;
    int lane    = (idx >> 4) & 31;
    int tileIdx = idx >> 9;                 // colTile*4 + ks
    int ks      = tileIdx & 3;
    int colTile = tileIdx >> 2;
    int col = colTile * 16 + (lane & 15);
    int k   = ks * 32 + (j >> 3) * 16 + (lane >> 4) * 8 + (j & 7);
    Wp[idx] = f2bf(W[(size_t)k * NCOL + col]);
}

// load + (optional add) + convert one 16x32 bf16 A fragment for this lane
template<bool ADD2>
__device__ __forceinline__ void load_a_frag(BFVec& a, const float* __restrict__ Ar,
                                            const float* __restrict__ A2r,
                                            int k0, int khalf) {
    const float4* p0 = (const float4*)(Ar + k0 + 8 * khalf);
    const float4* p1 = (const float4*)(Ar + k0 + 16 + 8 * khalf);
    float4 x0 = p0[0], x1 = p0[1], x2 = p1[0], x3 = p1[1];
    if (ADD2) {
        const float4* q0 = (const float4*)(A2r + k0 + 8 * khalf);
        const float4* q1 = (const float4*)(A2r + k0 + 16 + 8 * khalf);
        float4 y0 = q0[0], y1 = q0[1], y2 = q1[0], y3 = q1[1];
        x0.x += y0.x; x0.y += y0.y; x0.z += y0.z; x0.w += y0.w;
        x1.x += y1.x; x1.y += y1.y; x1.z += y1.z; x1.w += y1.w;
        x2.x += y2.x; x2.y += y2.y; x2.z += y2.z; x2.w += y2.w;
        x3.x += y3.x; x3.y += y3.y; x3.z += y3.z; x3.w += y3.w;
    }
    cvt4(&a.u[0],  x0);
    cvt4(&a.u[4],  x1);
    cvt4(&a.u[8],  x2);
    cvt4(&a.u[12], x3);
}

// ---------------------------------------------------------------------------
// WMMA GEMM: C[M,NCOL] = act(((A (+A2)) @ W) + bias), K = 128.
// Block = 128 threads (4 waves). Each block covers 32 rows (2 strips of 16);
// each wave computes NCOL/64 col-tiles (colTile = wave + 4t) for BOTH strips,
// reusing each loaded B tile. grid = (ceil(M/32), 1).
// ---------------------------------------------------------------------------
template<bool RELU, bool ADD2, int NCOL>
__global__ __launch_bounds__(128)
void gemm_wmma_k128(const float* __restrict__ A, const float* __restrict__ A2,
                    const unsigned short* __restrict__ Wp, const float* __restrict__ bias,
                    float* __restrict__ C, int M) {
    constexpr int K      = 128;
    constexpr int KSTEPS = K / 32;          // 4
    constexpr int TPW    = NCOL / 64;       // col tiles per wave: 2 (NCOL=128) or 1 (64)
    const int lane  = threadIdx.x & 31;
    const int wave  = threadIdx.x >> 5;
    const int mrow  = lane & 15;
    const int khalf = lane >> 4;

    const int rowBase0 = blockIdx.x << 5;
    const bool has2    = (rowBase0 + 16) < M;
    const int rowBase1 = has2 ? rowBase0 + 16 : rowBase0;

    const float* Ar0  = A + (size_t)(rowBase0 + mrow) * K;
    const float* Ar1  = A + (size_t)(rowBase1 + mrow) * K;
    const float* A2r0 = ADD2 ? (A2 + (size_t)(rowBase0 + mrow) * K) : nullptr;
    const float* A2r1 = ADD2 ? (A2 + (size_t)(rowBase1 + mrow) * K) : nullptr;

    v8f z = {};
    v8f acc[2][TPW];
#pragma unroll
    for (int s = 0; s < 2; ++s)
#pragma unroll
        for (int t = 0; t < TPW; ++t) acc[s][t] = z;

#pragma unroll
    for (int ks = 0; ks < KSTEPS; ++ks) {
        const int k0 = ks * 32;
        BFVec a0, a1;
        load_a_frag<ADD2>(a0, Ar0, A2r0, k0, khalf);
        load_a_frag<ADD2>(a1, Ar1, A2r1, k0, khalf);

#pragma unroll
        for (int t = 0; t < TPW; ++t) {
            const int colTile = wave + 4 * t;
            BFVec b;
            const uint4* bp = (const uint4*)Wp + ((size_t)(colTile * KSTEPS + ks) * 32 + lane) * 2;
            b.q[0] = bp[0];
            b.q[1] = bp[1];
            acc[0][t] = __builtin_amdgcn_wmma_f32_16x16x32_bf16(false, a0.v, false, b.v,
                                                                (short)0, acc[0][t], false, false);
            acc[1][t] = __builtin_amdgcn_wmma_f32_16x16x32_bf16(false, a1.v, false, b.v,
                                                                (short)0, acc[1][t], false, false);
        }
    }

#pragma unroll
    for (int s = 0; s < 2; ++s) {
        if (s == 1 && !has2) break;
        const int rb = s ? rowBase1 : rowBase0;
#pragma unroll
        for (int t = 0; t < TPW; ++t) {
            const int col = (wave + 4 * t) * 16 + mrow;
            const float bc = bias[col];
#pragma unroll
            for (int r = 0; r < 8; ++r) {
                float v = acc[s][t][r] + bc;
                if (RELU) v = fmaxf(v, 0.0f);
                C[(size_t)(rb + r + 8 * khalf) * NCOL + col] = v;
            }
        }
    }
}

// ---------------------------------------------------------------------------
// conv1 message: m = relu(x[src] + edge_attr@We1 + be1) scatter-summed (dim 7)
// ---------------------------------------------------------------------------
__global__ void msg_scatter7(const float* __restrict__ x, const float* __restrict__ ea,
                             const int* __restrict__ src, const int* __restrict__ dst,
                             const float* __restrict__ We, const float* __restrict__ be,
                             float* __restrict__ agg) {
    int t = blockIdx.x * blockDim.x + threadIdx.x;
    int e = t >> 3, f = t & 7;
    if (e >= N_EDGES || f >= 7) return;
    float a0 = ea[e * 3 + 0], a1 = ea[e * 3 + 1], a2 = ea[e * 3 + 2];
    int s = src[e], d = dst[e];
    float v = x[s * 7 + f] + a0 * We[f] + a1 * We[7 + f] + a2 * We[14 + f] + be[f];
    atomicAdd(&agg[d * 7 + f], fmaxf(v, 0.0f));
}

// conv2/3 message, dim 128, edge embedding (3->128) fused in.
__global__ void msg_scatter128(const float* __restrict__ h, const float* __restrict__ ea,
                               const int* __restrict__ src, const int* __restrict__ dst,
                               const float* __restrict__ We, const float* __restrict__ be,
                               float* __restrict__ agg) {
    int t = blockIdx.x * blockDim.x + threadIdx.x;
    int e = t >> 7, f = t & 127;
    if (e >= N_EDGES) return;
    float a0 = ea[e * 3 + 0], a1 = ea[e * 3 + 1], a2 = ea[e * 3 + 2];
    int s = src[e], d = dst[e];
    float v = h[(size_t)s * HID + f]
            + a0 * We[f] + a1 * We[HID + f] + a2 * We[2 * HID + f] + be[f];
    atomicAdd(&agg[(size_t)d * HID + f], fmaxf(v, 0.0f));
}

// conv1 lin1: out = relu((x+agg7) @ W[7,128] + b)
__global__ void lin7_relu(const float* __restrict__ x, const float* __restrict__ agg,
                          const float* __restrict__ W, const float* __restrict__ b,
                          float* __restrict__ out) {
    int t = blockIdx.x * blockDim.x + threadIdx.x;
    int r = t >> 7, c = t & 127;
    if (r >= N_NODES) return;
    float acc = b[c];
#pragma unroll
    for (int k = 0; k < 7; ++k)
        acc += (x[r * 7 + k] + agg[r * 7 + k]) * W[k * HID + c];
    out[(size_t)r * HID + c] = fmaxf(acc, 0.0f);
}

// BN column stats: partial sum / sumsq, 128 threads per block (coalesced rows)
__global__ void bn_stats(const float* __restrict__ h, float* __restrict__ stats) {
    int f = threadIdx.x;
    float s = 0.0f, s2 = 0.0f;
    for (int r = blockIdx.x; r < N_NODES; r += gridDim.x) {
        float v = h[(size_t)r * HID + f];
        s += v; s2 += v * v;
    }
    atomicAdd(&stats[f], s);
    atomicAdd(&stats[HID + f], s2);
}

__global__ void bn_apply_relu(float* __restrict__ h, const float* __restrict__ stats,
                              const float* __restrict__ g, const float* __restrict__ bt) {
    int t = blockIdx.x * blockDim.x + threadIdx.x;
    int r = t >> 7, c = t & 127;
    if (r >= N_NODES) return;
    const float inv = 1.0f / (float)N_NODES;
    float mean = stats[c] * inv;
    float var  = stats[HID + c] * inv - mean * mean;
    float sc   = g[c] * rsqrtf(var + BN_EPS);
    float v    = (h[t] - mean) * sc + bt[c];
    h[t] = fmaxf(v, 0.0f);
}

__global__ void pool_scatter(const float* __restrict__ h, const int* __restrict__ batch,
                             float* __restrict__ psum, float* __restrict__ pcnt) {
    int t = blockIdx.x * blockDim.x + threadIdx.x;
    int r = t >> 7, f = t & 127;
    if (r >= N_NODES) return;
    int gidx = batch[r];
    atomicAdd(&psum[(size_t)gidx * HID + f], h[t]);
    if (f == 0) atomicAdd(&pcnt[gidx], 1.0f);
}

__global__ void pool_div(const float* __restrict__ psum, const float* __restrict__ pcnt,
                         float* __restrict__ pooled) {
    int t = blockIdx.x * blockDim.x + threadIdx.x;
    if (t >= N_GRAPHS * HID) return;
    pooled[t] = psum[t] / fmaxf(pcnt[t >> 7], 1.0f);
}

__global__ void head_final(const float* __restrict__ h1, const float* __restrict__ Wf2,
                           const float* __restrict__ bf2, float* __restrict__ out) {
    int g = blockIdx.x * blockDim.x + threadIdx.x;
    if (g >= N_GRAPHS) return;
    float acc = bf2[0];
#pragma unroll
    for (int k = 0; k < 64; ++k) acc += h1[g * 64 + k] * Wf2[k];
    out[g] = acc;
}

// ---------------------------------------------------------------------------
extern "C" void kernel_launch(void* const* d_in, const int* in_sizes, int n_in,
                              void* d_out, int out_size, void* d_ws, size_t ws_size,
                              hipStream_t stream) {
    const float* x     = (const float*)d_in[0];
    const float* ea    = (const float*)d_in[1];
    const int*   eidx  = (const int*)d_in[2];
    const int*   src   = eidx;
    const int*   dst   = eidx + N_EDGES;
    const int*   batch = (const int*)d_in[3];
    const float *We1 = (const float*)d_in[4],  *be1 = (const float*)d_in[5];
    const float *W11 = (const float*)d_in[6],  *b11 = (const float*)d_in[7];
    const float *W12 = (const float*)d_in[8],  *b12 = (const float*)d_in[9];
    const float *g1  = (const float*)d_in[10], *bt1 = (const float*)d_in[11];
    const float *We2 = (const float*)d_in[12], *be2 = (const float*)d_in[13];
    const float *W21 = (const float*)d_in[14], *b21 = (const float*)d_in[15];
    const float *W22 = (const float*)d_in[16], *b22 = (const float*)d_in[17];
    const float *g2  = (const float*)d_in[18], *bt2 = (const float*)d_in[19];
    const float *We3 = (const float*)d_in[20], *be3 = (const float*)d_in[21];
    const float *W31 = (const float*)d_in[22], *b31 = (const float*)d_in[23];
    const float *W32 = (const float*)d_in[24], *b32 = (const float*)d_in[25];
    const float *g3  = (const float*)d_in[26], *bt3 = (const float*)d_in[27];
    const float *Wf1 = (const float*)d_in[28], *bf1 = (const float*)d_in[29];
    const float *Wf2 = (const float*)d_in[30], *bf2 = (const float*)d_in[31];
    float* out = (float*)d_out;

    float* ws = (float*)d_ws;
    const size_t S = (size_t)N_NODES * HID;
    float* bufA   = ws;
    float* bufB   = ws + S;
    float* agg    = ws + 2 * S;
    float* agg7   = ws + 3 * S;
    float* stats  = agg7 + (size_t)N_NODES * 7;
    float* psum   = stats + 2 * HID;
    float* pcnt   = psum + (size_t)N_GRAPHS * HID;
    float* pooled = pcnt + N_GRAPHS;
    float* head1  = pooled + (size_t)N_GRAPHS * HID;
    // packed bf16 weights (ushort), in float-unit slots after head1
    unsigned short* pW12 = (unsigned short*)(head1 + (size_t)N_GRAPHS * 64);
    unsigned short* pW21 = pW12 + 128 * 128;
    unsigned short* pW22 = pW21 + 128 * 128;
    unsigned short* pW31 = pW22 + 128 * 128;
    unsigned short* pW32 = pW31 + 128 * 128;
    unsigned short* pWf1 = pW32 + 128 * 128;

    const dim3 blk256(256), blk128(128);
    const dim3 gNode((N_NODES * HID) / 256);          // 25000
    const dim3 gEdge128((N_EDGES * HID) / 256);       // 400000
    const dim3 gEdge7((N_EDGES * 8) / 256);           // 25000
    const dim3 gGemm((N_NODES / 16 + 1) / 2, 1);      // 1563 blocks x 32 rows
    const dim3 gHead(N_GRAPHS / 32, 1);               // 64 blocks x 32 rows

    // ---- pack weights to bf16 WMMA layout (tiny; once per launch) ----
    pack_w_bf16<128><<<64, blk256, 0, stream>>>(W12, pW12);
    pack_w_bf16<128><<<64, blk256, 0, stream>>>(W21, pW21);
    pack_w_bf16<128><<<64, blk256, 0, stream>>>(W22, pW22);
    pack_w_bf16<128><<<64, blk256, 0, stream>>>(W31, pW31);
    pack_w_bf16<128><<<64, blk256, 0, stream>>>(W32, pW32);
    pack_w_bf16<64><<<32, blk256, 0, stream>>>(Wf1, pWf1);

    // ---- conv1 ----
    hipMemsetAsync(agg7, 0, (size_t)N_NODES * 7 * sizeof(float), stream);
    msg_scatter7<<<gEdge7, blk256, 0, stream>>>(x, ea, src, dst, We1, be1, agg7);
    lin7_relu<<<gNode, blk256, 0, stream>>>(x, agg7, W11, b11, bufA);
    gemm_wmma_k128<false, false, HID><<<gGemm, blk128, 0, stream>>>(bufA, nullptr, pW12, b12, bufB, N_NODES);
    hipMemsetAsync(stats, 0, 2 * HID * sizeof(float), stream);
    bn_stats<<<512, 128, 0, stream>>>(bufB, stats);
    bn_apply_relu<<<gNode, blk256, 0, stream>>>(bufB, stats, g1, bt1);

    // ---- conv2 ----
    hipMemsetAsync(agg, 0, S * sizeof(float), stream);
    msg_scatter128<<<gEdge128, blk256, 0, stream>>>(bufB, ea, src, dst, We2, be2, agg);
    gemm_wmma_k128<true, true, HID><<<gGemm, blk128, 0, stream>>>(bufB, agg, pW21, b21, bufA, N_NODES);
    gemm_wmma_k128<false, false, HID><<<gGemm, blk128, 0, stream>>>(bufA, nullptr, pW22, b22, bufB, N_NODES);
    hipMemsetAsync(stats, 0, 2 * HID * sizeof(float), stream);
    bn_stats<<<512, 128, 0, stream>>>(bufB, stats);
    bn_apply_relu<<<gNode, blk256, 0, stream>>>(bufB, stats, g2, bt2);

    // ---- conv3 ----
    hipMemsetAsync(agg, 0, S * sizeof(float), stream);
    msg_scatter128<<<gEdge128, blk256, 0, stream>>>(bufB, ea, src, dst, We3, be3, agg);
    gemm_wmma_k128<true, true, HID><<<gGemm, blk128, 0, stream>>>(bufB, agg, pW31, b31, bufA, N_NODES);
    gemm_wmma_k128<false, false, HID><<<gGemm, blk128, 0, stream>>>(bufA, nullptr, pW32, b32, bufB, N_NODES);
    hipMemsetAsync(stats, 0, 2 * HID * sizeof(float), stream);
    bn_stats<<<512, 128, 0, stream>>>(bufB, stats);
    bn_apply_relu<<<gNode, blk256, 0, stream>>>(bufB, stats, g3, bt3);

    // ---- global mean pool + head ----
    hipMemsetAsync(psum, 0, ((size_t)N_GRAPHS * HID + N_GRAPHS) * sizeof(float), stream);
    pool_scatter<<<gNode, blk256, 0, stream>>>(bufB, batch, psum, pcnt);
    pool_div<<<(N_GRAPHS * HID) / 256, blk256, 0, stream>>>(psum, pcnt, pooled);
    gemm_wmma_k128<true, false, 64><<<gHead, blk128, 0, stream>>>(pooled, nullptr, pWf1, bf1, head1, N_GRAPHS);
    head_final<<<(N_GRAPHS + 255) / 256, blk256, 0, stream>>>(head1, Wf2, bf2, out);
}
